// lstm_sig_10746008175157
// MI455X (gfx1250) — matrix operand order repaired and checked by reading, not compiled, not run on hardware
//
#include <hip/hip_runtime.h>

#define B_   256
#define L_   512
#define I_   32
#define H_   16
#define G_   64      // 4*H gates
#define SIG_ 4368    // 16 + 256 + 4096

typedef __attribute__((ext_vector_type(16))) _Float16 v16h;
typedef __attribute__((ext_vector_type(8)))  float    v8f;

// ---------------------------------------------------------------------------
// Phase 1: XW[b*L+l, g] = sum_i X[b,l,i] * W_ih[g,i] + b_ih[g] + b_hh[g]
// M = B*L = 131072, K = 32 (one 16x16x32 f16 WMMA), N = 64 (4 n-tiles).
// One wave per 16-row M tile; 8 waves per block.
// ---------------------------------------------------------------------------
__global__ __launch_bounds__(256)
void xw_wmma_kernel(const float* __restrict__ X,
                    const float* __restrict__ W_ih,
                    const float* __restrict__ b_ih,
                    const float* __restrict__ b_hh,
                    float* __restrict__ XW) {
  const int lane  = threadIdx.x & 31;
  const int wave  = threadIdx.x >> 5;
  const int mtile = blockIdx.x * 8 + wave;
  const int row0  = mtile * 16;
  const int khalf = lane >> 4;      // which K-half this lane holds
  const int l15   = lane & 15;      // M (for A) / N (for B,C)

  // A: 16x32 f16 tile of X.  Contiguous K order: lanes 0-15 hold M=0..15 with
  // K = 16*khalf + 0..15.
  v16h a;
  const float* xrow = X + (size_t)(row0 + l15) * I_;
  #pragma unroll
  for (int i = 0; i < 8; ++i) a[i]     = (_Float16)xrow[khalf * 16 + i];
  #pragma unroll
  for (int i = 0; i < 8; ++i) a[8 + i] = (_Float16)xrow[khalf * 16 + 8 + i];

  #pragma unroll
  for (int nt = 0; nt < 4; ++nt) {
    const int g = nt * 16 + l15;                 // gate column
    // B: 32x16 f16 (B[k][n] = W_ih[n][k]); lane n holds K = 16*khalf + j.
    const float* wrow = W_ih + g * I_ + khalf * 16;
    v16h bm;
    #pragma unroll
    for (int j = 0; j < 16; ++j) bm[j] = (_Float16)wrow[j];

    v8f c = {};
    c = __builtin_amdgcn_wmma_f32_16x16x32_f16(
        /*neg_a=*/false, a, /*neg_b=*/false, bm,
        /*c_mod=*/(short)0, c, /*reuse_a=*/false, /*reuse_b=*/false);

    const float bias = b_ih[g] + b_hh[g];        // per-column bias
    #pragma unroll
    for (int r = 0; r < 8; ++r) {
      const int row = row0 + r + khalf * 8;      // C/D layout
      XW[(size_t)row * G_ + g] = c[r] + bias;
    }
  }
}

// ---------------------------------------------------------------------------
// Phase 2: fused LSTM recurrence + depth-3 signature scan.
// One block (256 threads) per batch element. Wave 0 runs the LSTM cell with
// wave32 shuffles; thread ij (0..255) owns S2[ij] (1 reg) and
// S3[ij*16..ij*16+15] (16 regs).  Serial-path optimizations:
//   * XW loads software-pipelined one step ahead (+ global_prefetch 4 ahead)
//   * dx/S1old broadcast double-buffered by t&1 -> ONE barrier per step
// ---------------------------------------------------------------------------
__device__ __forceinline__ float sigm(float x) { return 1.f / (1.f + __expf(-x)); }

__global__ __launch_bounds__(256)
void lstm_sig_kernel(const float* __restrict__ XW,
                     const float* __restrict__ W_hh,
                     float* __restrict__ sig) {
  const int b    = blockIdx.x;
  const int tid  = threadIdx.x;
  const int lane = tid & 31;
  const int wave = tid >> 5;

  __shared__ float ldsDx[2][16];
  __shared__ float ldsS1[2][16];

  // per-thread signature state
  const int ij = tid;                // flat (i,j) of S2 / S3 row
  const int ii = ij >> 4, jj = ij & 15;
  float s2 = 0.f;
  float s3[16];
  #pragma unroll
  for (int k = 0; k < 16; ++k) s3[k] = 0.f;

  // wave-0 LSTM state: lane k (0..15) holds h[k], c[k], S1[k];
  // lane handles gates {lane, lane+32}.
  float h = 0.f, cc = 0.f, s1 = 0.f;
  float whh0[16], whh1[16];
  float xw0 = 0.f, xw1 = 0.f;                    // pipelined gate pre-acts
  const float* xwb = XW + (size_t)b * L_ * G_;
  if (wave == 0) {
    #pragma unroll
    for (int k = 0; k < 16; ++k) {
      whh0[k] = W_hh[lane * H_ + k];
      whh1[k] = W_hh[(lane + 32) * H_ + k];
    }
    xw0 = xwb[lane];                             // t = 0 row
    xw1 = xwb[32 + lane];
  }

  for (int t = 0; t < L_; ++t) {
    const int p = t & 1;
    float hnew = 0.f, dxv = 0.f;
    if (wave == 0) {
      float g0 = xw0, g1 = xw1;
      if (t + 1 < L_) {                          // issue next row's loads now
        xw0 = xwb[(t + 1) * G_ + lane];
        xw1 = xwb[(t + 1) * G_ + 32 + lane];
      }
      if (t + 4 < L_)                            // global_prefetch_b8 ahead
        __builtin_prefetch(&xwb[(t + 4) * G_ + lane], 0, 3);
      #pragma unroll
      for (int k = 0; k < 16; ++k) {
        const float hk = __shfl(h, k, 32);
        g0 += hk * whh0[k];
        g1 += hk * whh1[k];
      }
      // lane k<16: i=g0, g=g1; f lives in lane k+16 (g0), o in lane k+16 (g1)
      const float fg = __shfl(g0, (lane + 16) & 31, 32);
      const float og = __shfl(g1, (lane + 16) & 31, 32);
      cc   = sigm(fg) * cc + sigm(g0) * tanhf(g1);
      hnew = sigm(og) * tanhf(cc);
      dxv  = hnew - h;                           // path increment (lanes 0-15)
      if (lane < 16) { ldsDx[p][lane] = dxv; ldsS1[p][lane] = s1; }
    }
    __syncthreads();   // publish buf[p]; also guards buf[p] rewrite at t+2

    if (t >= 1) {
      float d[16];
      #pragma unroll
      for (int k = 0; k < 16; ++k) d[k] = ldsDx[p][k];
      const float s1i  = ldsS1[p][ii];
      // S3[ij,k] += (S2old[ij] + S1old[i]*d[j]/2 + d[i]*d[j]/6) * d[k]
      const float coef = s2 + 0.5f * s1i * d[jj] + (1.f / 6.f) * d[ii] * d[jj];
      #pragma unroll
      for (int k = 0; k < 16; ++k) s3[k] += coef * d[k];
      // S2[ij] += S1old[i]*d[j] + d[i]*d[j]/2
      s2 += s1i * d[jj] + 0.5f * d[ii] * d[jj];
    }
    if (wave == 0) {
      h = hnew;
      if (t >= 1 && lane < 16) s1 += dxv;        // S1 += dx
    }
  }

  // write concat(S1, S2, S3) for this batch element
  float* sb = sig + (size_t)b * SIG_;
  if (wave == 0 && lane < 16) sb[lane] = s1;
  sb[16 + ij] = s2;
  #pragma unroll
  for (int k = 0; k < 16; ++k) sb[16 + 256 + ij * 16 + k] = s3[k];
}

// ---------------------------------------------------------------------------
// Phase 3: out[b,o] = sig[b,:] . fc_w[o,:] + fc_b[o]
// One block per batch element; wave per output; shuffle reduction (wave32).
// ---------------------------------------------------------------------------
__global__ __launch_bounds__(320)
void fc_kernel(const float* __restrict__ sig,
               const float* __restrict__ fc_w,
               const float* __restrict__ fc_b,
               float* __restrict__ out) {
  const int b    = blockIdx.x;
  const int o    = threadIdx.x >> 5;   // 0..9
  const int lane = threadIdx.x & 31;
  const float* sb = sig + (size_t)b * SIG_;
  const float* wo = fc_w + (size_t)o * SIG_;
  float acc = 0.f;
  for (int d = lane; d < SIG_; d += 32) acc += sb[d] * wo[d];
  #pragma unroll
  for (int off = 16; off >= 1; off >>= 1) acc += __shfl_down(acc, off, 32);
  if (lane == 0) out[b * 10 + o] = acc + fc_b[o];
}

// ---------------------------------------------------------------------------
extern "C" void kernel_launch(void* const* d_in, const int* in_sizes, int n_in,
                              void* d_out, int out_size, void* d_ws, size_t ws_size,
                              hipStream_t stream) {
  const float* X    = (const float*)d_in[0];
  const float* W_ih = (const float*)d_in[1];
  const float* W_hh = (const float*)d_in[2];
  const float* b_ih = (const float*)d_in[3];
  const float* b_hh = (const float*)d_in[4];
  const float* fc_w = (const float*)d_in[5];
  const float* fc_b = (const float*)d_in[6];
  float* out = (float*)d_out;

  float* XW     = (float*)d_ws;                    // B*L*64 f32 = 32 MB
  float* sigbuf = XW + (size_t)B_ * L_ * G_;       // B*4368 f32

  const int mtiles = (B_ * L_) / 16;               // 8192
  xw_wmma_kernel<<<dim3(mtiles / 8), dim3(256), 0, stream>>>(X, W_ih, b_ih, b_hh, XW);
  lstm_sig_kernel<<<dim3(B_), dim3(256), 0, stream>>>(XW, W_hh, sigbuf);
  fc_kernel<<<dim3(B_), dim3(320), 0, stream>>>(sigbuf, fc_w, fc_b, out);
}
